// NeoformerMHA_84198538870852
// MI455X (gfx1250) — compile-verified, matrix operand order checked by reading, not run on hardware
//
#include <hip/hip_runtime.h>

// ---------------------------------------------------------------------------
// Neoformer MHA forward for gfx1250 (MI455X), bf16 WMMA + flash attention.
// B=4, F=512, T=2048, H=4, hf=128.
// ---------------------------------------------------------------------------

#define BB   4
#define FF   512
#define TT   2048
#define HH   4
#define HF   128

typedef __attribute__((ext_vector_type(16))) __bf16 bf16x16;
typedef __attribute__((ext_vector_type(8)))  __bf16 bf16x8;
typedef __attribute__((ext_vector_type(8)))  float  f32x8;

static __device__ __forceinline__ f32x8 wmma_bf16(bf16x16 a, bf16x16 b, f32x8 c) {
  return __builtin_amdgcn_wmma_f32_16x16x32_bf16(false, a, false, b, (short)0, c,
                                                 false, false);
}

// 32-bit LDS byte offset of a __shared__ object (addrspace(3) pointers are
// offsets from the workgroup LDS base).
static __device__ __forceinline__ unsigned lds_off(__bf16* p) {
  return (unsigned)(unsigned long long)(__attribute__((address_space(3))) __bf16*)p;
}

// Load a 16x32 bf16 fragment half for this lane.
// A-side: rowbase = &M-row[lane&15], K contiguous.  B-side: rowbase = &N-col
// row (stored contiguously in K).  Lane (n, hi): halves 0-7 = K hi*8..hi*8+7,
// halves 8-15 = K 16+hi*8..23+hi*8   (per ISA 16-bit 16x32 layout).
static __device__ __forceinline__ bf16x16 ldfrag(const __bf16* rowbase, int hi) {
  bf16x8 lo = *(const bf16x8*)(rowbase + hi * 8);
  bf16x8 up = *(const bf16x8*)(rowbase + 16 + hi * 8);
  bf16x16 a;
#pragma unroll
  for (int i = 0; i < 8; ++i) { a[i] = lo[i]; a[8 + i] = up[i]; }
  return a;
}

// Weight B-fragment: W row (one output column) is contiguous in f (=K dim).
static __device__ __forceinline__ bf16x16 ldfragW(const float* rowbase, int hi) {
  float4 f0 = *(const float4*)(rowbase + hi * 8);
  float4 f1 = *(const float4*)(rowbase + hi * 8 + 4);
  float4 f2 = *(const float4*)(rowbase + 16 + hi * 8);
  float4 f3 = *(const float4*)(rowbase + 16 + hi * 8 + 4);
  bf16x16 b;
  b[0] = (__bf16)f0.x; b[1] = (__bf16)f0.y; b[2]  = (__bf16)f0.z; b[3]  = (__bf16)f0.w;
  b[4] = (__bf16)f1.x; b[5] = (__bf16)f1.y; b[6]  = (__bf16)f1.z; b[7]  = (__bf16)f1.w;
  b[8] = (__bf16)f2.x; b[9] = (__bf16)f2.y; b[10] = (__bf16)f2.z; b[11] = (__bf16)f2.w;
  b[12] = (__bf16)f3.x; b[13] = (__bf16)f3.y; b[14] = (__bf16)f3.z; b[15] = (__bf16)f3.w;
  return b;
}

// ---------------------------------------------------------------------------
// Kernel 1: QKV projection + bias + RoPE.  1536 blocks x 128 threads.
// Per (b, 64-row t-supertile): 48 wave-tasks (0-15 Q, 16-31 K, 32-47 V).
// EVERY task computes two 16-col accumulator chains over four stacked 16-row
// M-subtiles -> completely uniform inner loop (no exec-masked WMMA), with 4x
// weight-traffic amortization via the shared LDS x-tile.
// ---------------------------------------------------------------------------
__global__ __launch_bounds__(128) void qkv_rope_kernel(
    const float* __restrict__ x,
    const float* __restrict__ Wq, const float* __restrict__ bq,
    const float* __restrict__ Wk, const float* __restrict__ bk,
    const float* __restrict__ Wv, const float* __restrict__ bv,
    __bf16* __restrict__ q1, __bf16* __restrict__ q2,
    __bf16* __restrict__ k1, __bf16* __restrict__ k2,
    __bf16* __restrict__ vt) {
  __shared__ __bf16 As[64 * 32];  // 4 KB: x supertile as bf16 [t(64)][k(32)]

  const int tid  = threadIdx.x;
  const int lane = tid & 31;
  const int wv   = tid >> 5;
  const int bt   = blockIdx.x / 12;
  const int sub  = blockIdx.x - bt * 12;
  const int b    = bt >> 5;
  const int ts0  = (bt & 31) << 6;   // 64-row supertile base
  const int task = sub * 4 + wv;
  const int n    = lane & 15;
  const int hi   = lane >> 4;

  const bool isV = (task >= 32);
  const float* Wm;
  const float* bias;
  int colR, colI, hh = 0, p = 0, cidx = 0;
  if (!isV) {
    Wm   = (task < 16) ? Wq : Wk;
    bias = (task < 16) ? bq : bk;
    const int tt = task & 15;
    hh = tt >> 2; p = (tt >> 1) & 1; cidx = tt & 1;
    colR = hh * HF + p * 64 + cidx * 16;
    colI = colR + 32;
  } else {
    Wm = Wv; bias = bv;
    colR = (task - 32) * 32;
    colI = colR + 16;
  }

  f32x8 accR[4], accI[4];
#pragma unroll
  for (int s = 0; s < 4; ++s)
#pragma unroll
    for (int i = 0; i < 8; ++i) { accR[s][i] = 0.f; accI[s][i] = 0.f; }

  for (int k0 = 0; k0 < FF; k0 += 32) {
    __syncthreads();
    // Stage x tile [k0..k0+31][ts0..ts0+63] as bf16 A-tile [t][k] in LDS.
    // float4 along t: 16 lanes cover one 64-float row (256B coalesced).
#pragma unroll
    for (int j = 0; j < 4; ++j) {
      const int c  = tid + j * 128;   // 0..511 chunks of 4 elems
      const int kk = c >> 4;          // k row 0..31
      const int t4 = (c & 15) * 4;    // t col 0,4,...,60
      const float4 v4 =
          *(const float4*)(x + ((size_t)b * FF + (k0 + kk)) * TT + ts0 + t4);
      As[(t4 + 0) * 32 + kk] = (__bf16)v4.x;
      As[(t4 + 1) * 32 + kk] = (__bf16)v4.y;
      As[(t4 + 2) * 32 + kk] = (__bf16)v4.z;
      As[(t4 + 3) * 32 + kk] = (__bf16)v4.w;
    }
    __syncthreads();

    const bf16x16 bRf = ldfragW(Wm + (size_t)(colR + n) * FF + k0, hi);
    const bf16x16 bIf = ldfragW(Wm + (size_t)(colI + n) * FF + k0, hi);

#pragma unroll
    for (int s = 0; s < 4; ++s) {
      const bf16x16 a = ldfrag(&As[(s * 16 + n) * 32], hi);
      accR[s] = wmma_bf16(a, bRf, accR[s]);
      accI[s] = wmma_bf16(a, bIf, accI[s]);
    }
  }

  if (!isV) {
    const float bR = bias[colR + n];
    const float bI = bias[colI + n];
    __bf16* ob = (task < 16) ? (p ? q2 : q1) : (p ? k2 : k1);
    const int j = cidx * 16 + n;
    const float invf = exp2f(-(float)j) * 0.0625f;  // 2^-j / 16
#pragma unroll
    for (int s = 0; s < 4; ++s) {
#pragma unroll
      for (int r = 0; r < 8; ++r) {
        const int t = ts0 + s * 16 + r + hi * 8;
        const float vr = accR[s][r] + bR;
        const float vi = accI[s][r] + bI;
        float sn, cs;
        __sincosf((float)t * invf, &sn, &cs);
        const size_t rowb = ((size_t)(b * HH + hh) * TT + t) * 64;
        ob[rowb + j]      = (__bf16)(vr * sn - vi * cs);  // rotated "real"
        ob[rowb + 32 + j] = (__bf16)(vr * cs + vi * sn);  // rotated "imag"
      }
    }
  } else {
#pragma unroll
    for (int part = 0; part < 2; ++part) {
      const int oc = (part ? colI : colR) + n;   // global feature
      const int h2 = oc >> 7;
      const int fo = oc & 127;
      const float bb = bias[oc];
#pragma unroll
      for (int s = 0; s < 4; ++s) {
        bf16x8 pk;
#pragma unroll
        for (int r = 0; r < 8; ++r)
          pk[r] = (__bf16)((part ? accI[s][r] : accR[s][r]) + bb);
        // V stored transposed: [B,H,128,T] so AV B-fragments are contiguous.
        const size_t base =
            ((size_t)(b * HH + h2) * HF + fo) * TT + ts0 + s * 16 + hi * 8;
        *(bf16x8*)(vt + base) = pk;
      }
    }
  }
}

// ---------------------------------------------------------------------------
// Async staging of one 32-key K/V tile into LDS (GLOBAL_LOAD_ASYNC_TO_LDS,
// tracked by ASYNCcnt; 6 async instructions per wave per tile).
// ---------------------------------------------------------------------------
static __device__ __forceinline__ void stage_tile_async(
    const __bf16* __restrict__ kws, const __bf16* __restrict__ vt,
    size_t bhT, size_t vbase, int j0, int tid,
    unsigned kdst, unsigned vdst) {
  // K tile [32 keys][64 f]: contiguous 2048 bf16 in kws.
  const unsigned long long ksrc =
      (unsigned long long)(const void*)(kws + (bhT + j0) * 64);
#pragma unroll
  for (int j = 0; j < 2; ++j) {
    const int c = tid + j * 128;
    asm volatile("global_load_async_to_lds_b128 %0, %1, off"
                 :: "v"(kdst + c * 16), "v"(ksrc + (unsigned long long)c * 16)
                 : "memory");
  }
  // V tile [128 f][32 keys] gathered from transposed V rows.
#pragma unroll
  for (int j = 0; j < 4; ++j) {
    const int c  = tid + j * 128;
    const int f  = c >> 2;
    const int kc = c & 3;
    const unsigned long long vsrc = (unsigned long long)(const void*)(
        vt + vbase + (size_t)f * TT + j0 + kc * 8);
    asm volatile("global_load_async_to_lds_b128 %0, %1, off"
                 :: "v"(vdst + c * 16), "v"(vsrc)
                 : "memory");
  }
}

// ---------------------------------------------------------------------------
// Flash-attention pass: softmax(Q*K^T/16) @ V for one 16-query tile per wave.
// The 4 waves of a block share (b,h); K/V tiles are DOUBLE-BUFFERED in LDS
// via async global->LDS copies (issue next tile, s_wait_asynccnt 6 for the
// current one, barrier, compute).
// ---------------------------------------------------------------------------
static __device__ __forceinline__ void flash_pass(
    const __bf16* __restrict__ qws, const __bf16* __restrict__ kws,
    const __bf16* __restrict__ vt, __bf16* Ks, __bf16* Vs, __bf16* pl,
    unsigned koff, unsigned voff,
    size_t bhT, size_t vbase, int q0, int lane, int tid, f32x8 o[8]) {
  const int n  = lane & 15;
  const int hi = lane >> 4;

  const __bf16* qrow = qws + (bhT + (size_t)(q0 + n)) * 64;
  const bf16x16 qa0 = ldfrag(qrow, hi);
  const bf16x16 qa1 = ldfrag(qrow + 32, hi);

  float mr[8], lr[8];
#pragma unroll
  for (int r = 0; r < 8; ++r) {
    mr[r] = -3.0e38f; lr[r] = 0.f;
#pragma unroll
    for (int i = 0; i < 8; ++i) o[r][i] = 0.f;
  }

  // Prologue: stage tile 0 into buffer 0.
  stage_tile_async(kws, vt, bhT, vbase, 0, tid, koff, voff);

  for (int j0 = 0; j0 < TT; j0 += 32) {
    const int cur = (j0 >> 5) & 1;
    if (j0 + 32 < TT) {
      // Refill the other buffer (WAR covered by last iteration's barrier),
      // then wait so that only the 6 just-issued ops remain outstanding.
      stage_tile_async(kws, vt, bhT, vbase, j0 + 32, tid,
                       koff + (cur ^ 1) * 4096, voff + (cur ^ 1) * 8192);
      asm volatile("s_wait_asynccnt 0x6" ::: "memory");
    } else {
      asm volatile("s_wait_asynccnt 0x0" ::: "memory");
    }
    __syncthreads();   // every wave's share of the current tile has landed

    const __bf16* kb = Ks + cur * (32 * 64);
    const __bf16* vb = Vs + cur * (128 * 32);

    // --- S = Q * K^T (two 16x16 key subtiles). ---
    f32x8 slo, shi;
#pragma unroll
    for (int i = 0; i < 8; ++i) { slo[i] = 0.f; shi[i] = 0.f; }
    const __bf16* kr0 = kb + n * 64;
    const __bf16* kr1 = kb + (16 + n) * 64;
    slo = wmma_bf16(qa0, ldfrag(kr0, hi), slo);
    slo = wmma_bf16(qa1, ldfrag(kr0 + 32, hi), slo);
    shi = wmma_bf16(qa0, ldfrag(kr1, hi), shi);
    shi = wmma_bf16(qa1, ldfrag(kr1 + 32, hi), shi);

    // --- Online softmax (rows live across 16-lane groups). ---
#pragma unroll
    for (int r = 0; r < 8; ++r) {
      const float sl = slo[r] * 0.0625f;    // /sqrt(256)
      const float sh = shi[r] * 0.0625f;
      float rm = fmaxf(sl, sh);
      rm = fmaxf(rm, __shfl_xor(rm, 1, 32));
      rm = fmaxf(rm, __shfl_xor(rm, 2, 32));
      rm = fmaxf(rm, __shfl_xor(rm, 4, 32));
      rm = fmaxf(rm, __shfl_xor(rm, 8, 32));
      const float mn = fmaxf(mr[r], rm);
      const float al = __expf(mr[r] - mn);
      const float p0 = __expf(sl - mn);
      const float p1 = __expf(sh - mn);
      float sum = p0 + p1;
      sum += __shfl_xor(sum, 1, 32);
      sum += __shfl_xor(sum, 2, 32);
      sum += __shfl_xor(sum, 4, 32);
      sum += __shfl_xor(sum, 8, 32);
      lr[r] = lr[r] * al + sum;
      mr[r] = mn;
#pragma unroll
      for (int nt = 0; nt < 8; ++nt) o[nt][r] *= al;
      const int row = r + hi * 8;           // C-layout -> LDS [q][k]
      pl[row * 32 + n]      = (__bf16)p0;
      pl[row * 32 + 16 + n] = (__bf16)p1;
    }

    // --- O += P * V  (P as A-fragment from LDS; V rows as B-fragments). ---
    const bf16x16 pa = ldfrag(pl + n * 32, hi);
#pragma unroll
    for (int nt = 0; nt < 8; ++nt) {
      o[nt] = wmma_bf16(pa, ldfrag(vb + (nt * 16 + n) * 32, hi), o[nt]);
    }

    __syncthreads();   // all waves done reading 'cur' before it is refilled
  }

#pragma unroll
  for (int r = 0; r < 8; ++r) {
    const float inv = 1.0f / lr[r];
#pragma unroll
    for (int nt = 0; nt < 8; ++nt) o[nt][r] *= inv;
  }
}

// ---------------------------------------------------------------------------
// Kernel 2: dual flash attention + residual + transpose-out.
// 512 blocks x 128 threads; one wave per (b,h,q-tile of 16); the 4 waves of
// a block share (b,h) and the double-buffered async-staged K/V tiles.
// ---------------------------------------------------------------------------
__global__ __launch_bounds__(128) void attn_kernel(
    const float* __restrict__ x, const float* __restrict__ s2,
    const __bf16* __restrict__ q1, const __bf16* __restrict__ q2,
    const __bf16* __restrict__ k1, const __bf16* __restrict__ k2,
    const __bf16* __restrict__ vt, float* __restrict__ out) {
  __shared__ __bf16 Ks[2][32 * 64];      // 8 KB  (double-buffered)
  __shared__ __bf16 Vs[2][128 * 32];     // 16 KB (double-buffered)
  __shared__ __bf16 Plds[4][16 * 32];    // 4 KB  (per-wave P bounce)

  const int tid  = threadIdx.x;
  const int lane = tid & 31;
  const int wv   = tid >> 5;
  const int gw   = blockIdx.x * 4 + wv;
  const int b    = gw >> 9;
  const int rem  = gw & 511;
  const int h    = rem >> 7;
  const int q0   = (rem & 127) << 4;
  const int n    = lane & 15;
  const int hi   = lane >> 4;

  const size_t bhT   = ((size_t)b * HH + h) * TT;
  const size_t vbase = ((size_t)b * HH + h) * HF * TT;
  const unsigned koff = lds_off(&Ks[0][0]);
  const unsigned voff = lds_off(&Vs[0][0]);

  f32x8 o1[8];
  flash_pass(q1, k1, vt, &Ks[0][0], &Vs[0][0], Plds[wv], koff, voff,
             bhT, vbase, q0, lane, tid, o1);

  // out = x + O1   (feature-major rows; 8 consecutive t per lane -> float4x2)
#pragma unroll
  for (int nt = 0; nt < 8; ++nt) {
    const size_t idx =
        ((size_t)b * FF + h * HF + nt * 16 + n) * TT + q0 + hi * 8;
    const float4 xa = *(const float4*)(x + idx);
    const float4 xb = *(const float4*)(x + idx + 4);
    float4 ya, yb;
    ya.x = xa.x + o1[nt][0]; ya.y = xa.y + o1[nt][1];
    ya.z = xa.z + o1[nt][2]; ya.w = xa.w + o1[nt][3];
    yb.x = xb.x + o1[nt][4]; yb.y = xb.y + o1[nt][5];
    yb.z = xb.z + o1[nt][6]; yb.w = xb.w + o1[nt][7];
    *(float4*)(out + idx)     = ya;
    *(float4*)(out + idx + 4) = yb;
  }

  f32x8 o2[8];
  flash_pass(q2, k2, vt, &Ks[0][0], &Vs[0][0], Plds[wv], koff, voff,
             bhT, vbase, q0, lane, tid, o2);

  const float s = s2[h];
#pragma unroll
  for (int nt = 0; nt < 8; ++nt) {
    const size_t idx =
        ((size_t)b * FF + h * HF + nt * 16 + n) * TT + q0 + hi * 8;
    float4 ya = *(const float4*)(out + idx);
    float4 yb = *(const float4*)(out + idx + 4);
    ya.x -= s * o2[nt][0]; ya.y -= s * o2[nt][1];
    ya.z -= s * o2[nt][2]; ya.w -= s * o2[nt][3];
    yb.x -= s * o2[nt][4]; yb.y -= s * o2[nt][5];
    yb.z -= s * o2[nt][6]; yb.w -= s * o2[nt][7];
    *(float4*)(out + idx)     = ya;
    *(float4*)(out + idx + 4) = yb;
  }
}

// ---------------------------------------------------------------------------
extern "C" void kernel_launch(void* const* d_in, const int* in_sizes, int n_in,
                              void* d_out, int out_size, void* d_ws,
                              size_t ws_size, hipStream_t stream) {
  (void)in_sizes; (void)n_in; (void)out_size; (void)ws_size;
  const float* x  = (const float*)d_in[0];
  const float* Wq = (const float*)d_in[1];
  const float* bq = (const float*)d_in[2];
  const float* Wk = (const float*)d_in[3];
  const float* bk = (const float*)d_in[4];
  const float* Wv = (const float*)d_in[5];
  const float* bv = (const float*)d_in[6];
  const float* s2 = (const float*)d_in[7];
  float* out = (float*)d_out;

  const size_t QK_ELEMS = (size_t)BB * HH * TT * 64;   // 2M bf16 each
  __bf16* q1 = (__bf16*)d_ws;
  __bf16* q2 = q1 + QK_ELEMS;
  __bf16* k1 = q2 + QK_ELEMS;
  __bf16* k2 = k1 + QK_ELEMS;
  __bf16* vt = k2 + QK_ELEMS;                          // [B,H,128,T]

  qkv_rope_kernel<<<dim3(BB * (TT / 64) * 12), dim3(128), 0, stream>>>(
      x, Wq, bq, Wk, bk, Wv, bv, q1, q2, k1, k2, vt);
  attn_kernel<<<dim3(BB * HH * (TT / 16) / 4), dim3(128), 0, stream>>>(
      x, s2, q1, q2, k1, k2, vt, out);
}
